// LSTM_68212670595845
// MI455X (gfx1250) — compile-verified, matrix operand order checked by reading, not run on hardware
//
#include <hip/hip_runtime.h>

// Problem dims (match reference)
#define TT 512
#define BB 64
#define DD 512
#define HH 1024
#define OO 512
#define KC (DD + HH)   // 1536 : concatenated reduction dim [x_t | h]
#define G4 (4 * HH)    // 4096 : fused gate dimension

typedef __attribute__((ext_vector_type(16))) __bf16 v16bf;
typedef __attribute__((ext_vector_type(8)))  float  v8f;
typedef unsigned int u32x4 __attribute__((ext_vector_type(4)));

union Frag {
    u32x4 q[2];   // two aligned 16B loads
    v16bf v;      // WMMA operand view
};

__device__ __forceinline__ float sigmoid_f(float x) {
    return 1.0f / (1.0f + __expf(-x));
}

// ---------------- conversion / setup kernels ----------------

__global__ void k_cvt_bf16(const float* __restrict__ src, __bf16* __restrict__ dst, int n) {
    int i = blockIdx.x * blockDim.x + threadIdx.x;
    if (i < n) dst[i] = (__bf16)src[i];
}

// Wcat[g, k] = k < D ? Wx[g,k] : Wh[g,k-D]   (row-major [4H, 1536], bf16)
__global__ void k_pack_w(const float* __restrict__ Wx, const float* __restrict__ Wh,
                         __bf16* __restrict__ Wcat) {
    int i = blockIdx.x * blockDim.x + threadIdx.x;
    if (i >= G4 * KC) return;
    int g = i / KC;
    int k = i - g * KC;
    float v = (k < DD) ? Wx[g * DD + k] : Wh[g * HH + (k - DD)];
    Wcat[i] = (__bf16)v;
}

__global__ void k_bsum(const float* __restrict__ bx, const float* __restrict__ bh,
                       float* __restrict__ bs) {
    int i = blockIdx.x * blockDim.x + threadIdx.x;
    if (i < G4) bs[i] = bx[i] + bh[i];
}

__global__ void k_init_state(float* __restrict__ c, __bf16* __restrict__ h) {
    int i = blockIdx.x * blockDim.x + threadIdx.x;
    if (i < BB * HH) { c[i] = 0.0f; h[i] = (__bf16)0.0f; }
}

// ---------------- fused LSTM step ----------------
// grid.x = HH/16 = 64 (hidden slice), block = 512 threads = 16 waves.
// wave = (m_tile 0..3) x (gate 0..3). Each wave: one 16x16 WMMA tile of
// gates = [x_t | h_prev] @ Wcat^T over K=1536 in bf16, f32 accumulate.
__global__ __launch_bounds__(512, 1)
void k_lstm_step(const __bf16* __restrict__ xt,     // [BB, DD] bf16 (timestep slice)
                 const __bf16* __restrict__ Wcat,   // [G4, KC] bf16 row-major
                 const float*  __restrict__ bsum,   // [G4]
                 float*        __restrict__ cstate, // [BB, HH] f32 (in/out)
                 const __bf16* __restrict__ h_in,   // [BB, HH] bf16
                 __bf16*       __restrict__ h_out)  // [BB, HH] bf16
{
    __shared__ float gl[4][BB][16];   // [gate][batch row][n] raw gate pre-activations

    const int lane  = threadIdx.x & 31;
    const int wv    = threadIdx.x >> 5;   // 0..15
    const int mt    = wv & 3;             // batch tile (rows 16*mt..16*mt+15)
    const int gate  = wv >> 2;            // 0..3 : i,f,o,c
    const int nlo   = lane & 15;
    const int khalf = (lane >> 4) * 8;    // hi lane half handles K+8 / K+24 (and M/N+8 for C/D)
    const int rowA  = mt * 16 + nlo;                           // A row (batch)
    const int colB  = gate * HH + blockIdx.x * 16 + nlo;       // W row (gate column)

    const __bf16* arow0 = xt   + (size_t)rowA * DD;
    const __bf16* arow1 = h_in + (size_t)rowA * HH;
    const __bf16* wrow  = Wcat + (size_t)colB * KC;

    v8f acc = {};

    // K = 0..511 : x_t projection
#pragma unroll 4
    for (int kb = 0; kb < DD; kb += 32) {
        Frag A, Bm;
        A.q[0]  = *(const u32x4*)(arow0 + kb + khalf);
        A.q[1]  = *(const u32x4*)(arow0 + kb + 16 + khalf);
        Bm.q[0] = *(const u32x4*)(wrow + kb + khalf);
        Bm.q[1] = *(const u32x4*)(wrow + kb + 16 + khalf);
        acc = __builtin_amdgcn_wmma_f32_16x16x32_bf16(
                  false, A.v, false, Bm.v, (short)0, acc, false, false);
    }
    // K = 512..1535 : recurrent h projection
#pragma unroll 4
    for (int kb = 0; kb < HH; kb += 32) {
        Frag A, Bm;
        A.q[0]  = *(const u32x4*)(arow1 + kb + khalf);
        A.q[1]  = *(const u32x4*)(arow1 + kb + 16 + khalf);
        Bm.q[0] = *(const u32x4*)(wrow + DD + kb + khalf);
        Bm.q[1] = *(const u32x4*)(wrow + DD + kb + 16 + khalf);
        acc = __builtin_amdgcn_wmma_f32_16x16x32_bf16(
                  false, A.v, false, Bm.v, (short)0, acc, false, false);
    }

    // C/D layout: VGPR i, lanes 0-15 -> M=i ; lanes 16-31 -> M=8+i ; N = lane&15
#pragma unroll
    for (int i = 0; i < 8; ++i)
        gl[gate][mt * 16 + khalf + i][nlo] = acc[i];

    __syncthreads();

    // Elementwise gate combine: 64 rows x 16 cols, 512 threads -> 2 elems each
    for (int e = threadIdx.x; e < BB * 16; e += 512) {
        int r = e >> 4;
        int n = e & 15;
        int j = blockIdx.x * 16 + n;          // hidden index 0..1023
        float gi = gl[0][r][n] + bsum[0 * HH + j];
        float gf = gl[1][r][n] + bsum[1 * HH + j];
        float go = gl[2][r][n] + bsum[2 * HH + j];
        float gc = gl[3][r][n] + bsum[3 * HH + j];
        float it = sigmoid_f(gi);
        float ft = sigmoid_f(gf);
        float ot = sigmoid_f(go);
        float ct = tanhf(gc);
        float cn = ft * cstate[(size_t)r * HH + j] + it * ct;
        cstate[(size_t)r * HH + j] = cn;
        h_out[(size_t)r * HH + j]  = (__bf16)(ot * tanhf(cn));
    }
}

// ---------------- final projection: out = h @ Why^T + bhy ----------------
// grid.x = OO/16 = 32, block = 128 threads = 4 waves (one per batch tile).
__global__ __launch_bounds__(128, 1)
void k_out_gemm(const __bf16* __restrict__ h,     // [BB, HH] bf16
                const __bf16* __restrict__ Wy,    // [OO, HH] bf16 row-major
                const float*  __restrict__ bhy,   // [OO]
                float*        __restrict__ out)   // [BB, OO] f32
{
    const int lane  = threadIdx.x & 31;
    const int mt    = threadIdx.x >> 5;   // 0..3
    const int nlo   = lane & 15;
    const int khalf = (lane >> 4) * 8;
    const int rowA  = mt * 16 + nlo;
    const int colB  = blockIdx.x * 16 + nlo;

    const __bf16* arow = h  + (size_t)rowA * HH;
    const __bf16* wrow = Wy + (size_t)colB * HH;

    v8f acc = {};
#pragma unroll 4
    for (int kb = 0; kb < HH; kb += 32) {
        Frag A, Bm;
        A.q[0]  = *(const u32x4*)(arow + kb + khalf);
        A.q[1]  = *(const u32x4*)(arow + kb + 16 + khalf);
        Bm.q[0] = *(const u32x4*)(wrow + kb + khalf);
        Bm.q[1] = *(const u32x4*)(wrow + kb + 16 + khalf);
        acc = __builtin_amdgcn_wmma_f32_16x16x32_bf16(
                  false, A.v, false, Bm.v, (short)0, acc, false, false);
    }

    float bias = bhy[colB];
#pragma unroll
    for (int i = 0; i < 8; ++i)
        out[(size_t)(mt * 16 + khalf + i) * OO + colB] = acc[i] + bias;
}

// ---------------- host launcher ----------------

extern "C" void kernel_launch(void* const* d_in, const int* in_sizes, int n_in,
                              void* d_out, int out_size, void* d_ws, size_t ws_size,
                              hipStream_t stream) {
    (void)in_sizes; (void)n_in; (void)out_size; (void)ws_size;

    const float* x   = (const float*)d_in[0];   // [T,B,D]
    const float* Wx  = (const float*)d_in[1];   // [4H,D]
    const float* bx  = (const float*)d_in[2];   // [4H]
    const float* Wh  = (const float*)d_in[3];   // [4H,H]
    const float* bh  = (const float*)d_in[4];   // [4H]
    const float* Why = (const float*)d_in[5];   // [O,H]
    const float* bhy = (const float*)d_in[6];   // [O]
    float* out = (float*)d_out;                 // [B,O]

    // Workspace carve-out (256B aligned slabs), ~47.7 MB total
    char* p = (char*)d_ws;
    auto carve = [&](size_t bytes) {
        char* r = p;
        p += (bytes + 255) & ~(size_t)255;
        return r;
    };
    __bf16* x_bf   = (__bf16*)carve((size_t)TT * BB * DD * 2);  // 33.6 MB
    __bf16* Wcat   = (__bf16*)carve((size_t)G4 * KC * 2);       // 12.6 MB (L2 resident)
    __bf16* Wy_bf  = (__bf16*)carve((size_t)OO * HH * 2);       //  1.0 MB
    float*  bsum   = (float*) carve((size_t)G4 * 4);
    float*  cstate = (float*) carve((size_t)BB * HH * 4);
    __bf16* hbuf0  = (__bf16*)carve((size_t)BB * HH * 2);
    __bf16* hbuf1  = (__bf16*)carve((size_t)BB * HH * 2);
    __bf16* hbuf[2] = { hbuf0, hbuf1 };

    // 1) convert / pack (bf16 once; weights then live in L2 across all steps)
    {
        int n = TT * BB * DD;
        k_cvt_bf16<<<(n + 255) / 256, 256, 0, stream>>>(x, x_bf, n);
    }
    {
        int n = G4 * KC;
        k_pack_w<<<(n + 255) / 256, 256, 0, stream>>>(Wx, Wh, Wcat);
    }
    {
        int n = OO * HH;
        k_cvt_bf16<<<(n + 255) / 256, 256, 0, stream>>>(Why, Wy_bf, n);
    }
    k_bsum<<<(G4 + 255) / 256, 256, 0, stream>>>(bx, bh, bsum);
    k_init_state<<<(BB * HH + 255) / 256, 256, 0, stream>>>(cstate, hbuf[0]);

    // 2) 512 fused recurrent steps (stream-ordered; h double-buffered)
    for (int t = 0; t < TT; ++t) {
        const __bf16* xt = x_bf + (size_t)t * BB * DD;
        k_lstm_step<<<HH / 16, 512, 0, stream>>>(
            xt, Wcat, bsum, cstate, hbuf[t & 1], hbuf[(t + 1) & 1]);
    }

    // 3) final projection from h_T (lands in hbuf[TT & 1] == hbuf[0])
    k_out_gemm<<<OO / 16, 128, 0, stream>>>(hbuf[TT & 1], Wy_bf, bhy, out);
}
